// SelfAttention_51419348468305
// MI455X (gfx1250) — matrix operand order repaired and checked by reading, not compiled, not run on hardware
//
#include <hip/hip_runtime.h>

typedef __bf16 bf16t;
typedef __attribute__((ext_vector_type(16))) __bf16 v16bf;
typedef __attribute__((ext_vector_type(8)))  __bf16 v8bf;
typedef __attribute__((ext_vector_type(8)))  float   v8f;

#define Qn  2048
#define Dn  2048
#define Hn  16
#define HDn 128
#define Pn  512
#define KVn (Pn + Qn)   // 2560

#ifndef __has_builtin
#define __has_builtin(x) 0
#endif

#if __has_builtin(__builtin_amdgcn_global_load_async_to_lds_b128)
#define HAVE_ASYNC 1
#else
#define HAVE_ASYNC 0
#endif

typedef int v4i_t __attribute__((vector_size(16)));
typedef __attribute__((address_space(1))) void*  as1p;
typedef __attribute__((address_space(3))) void*  as3p;
typedef __attribute__((address_space(1))) v4i_t* as1v4p;
typedef __attribute__((address_space(3))) v4i_t* as3v4p;

// Copy one 16-byte chunk global->LDS via the async engine (ASYNCcnt).
__device__ __forceinline__ void copy16_g2l(const bf16t* g, bf16t* l) {
#if HAVE_ASYNC
  __builtin_amdgcn_global_load_async_to_lds_b128(
      (as1v4p)(as1p)(void*)g, (as3v4p)(as3p)(void*)l, 0, 0);
#else
  *(v8bf*)l = *(const v8bf*)g;
#endif
}

// Wait until at most the 8 most recently issued async chunks are in flight:
// per-wave async loads complete in order, so this releases the previous tile
// while the next tile keeps streaming into LDS behind the WMMAs.
__device__ __forceinline__ void async_wait_prev8() {
#if HAVE_ASYNC
#if __has_builtin(__builtin_amdgcn_s_wait_asynccnt)
  __builtin_amdgcn_s_wait_asynccnt(8);
#else
  asm volatile("s_wait_asynccnt 0x8" ::: "memory");
#endif
#endif
}

__device__ __forceinline__ void async_join() {
#if HAVE_ASYNC
#if __has_builtin(__builtin_amdgcn_s_wait_asynccnt)
  __builtin_amdgcn_s_wait_asynccnt(0);
#else
  asm volatile("s_wait_asynccnt 0x0" ::: "memory");
#endif
#endif
}

// ---------------------------------------------------------------------------
// WMMA helpers (CDNA5 16x16x32 bf16, f32 accumulate)
// ---------------------------------------------------------------------------
__device__ __forceinline__ v8f wmma_bf16(v16bf a, v16bf b, v8f c) {
  return __builtin_amdgcn_wmma_f32_16x16x32_bf16(false, a, false, b,
                                                 (short)0, c, false, false);
}

// A fragment: 16x32 bf16 row-major in LDS.
__device__ __forceinline__ v16bf load_a_frag(const bf16t* base, int stride, int lane) {
  const int m  = lane & 15;
  const int kh = (lane >> 4) << 3;
  const bf16t* p = base + m * stride + kh;
  v8bf lo = *(const v8bf*)(p);
  v8bf hi = *(const v8bf*)(p + 16);
  return __builtin_shufflevector(lo, hi, 0,1,2,3,4,5,6,7,8,9,10,11,12,13,14,15);
}

// B fragment: 32x16 bf16 stored N-major (Bt[n][k]).
__device__ __forceinline__ v16bf load_b_frag(const bf16t* base, int stride, int lane) {
  const int n  = lane & 15;
  const int kh = (lane >> 4) << 3;
  const bf16t* p = base + n * stride + kh;
  v8bf lo = *(const v8bf*)(p);
  v8bf hi = *(const v8bf*)(p + 16);
  return __builtin_shufflevector(lo, hi, 0,1,2,3,4,5,6,7,8,9,10,11,12,13,14,15);
}

// ---------------------------------------------------------------------------
// One-time precision / layout conversion kernels
// ---------------------------------------------------------------------------
__global__ __launch_bounds__(256) void cvt_flat_kernel(
    const float* __restrict__ src, bf16t* __restrict__ dst, long n4)
{
  long i = (long)blockIdx.x * 256 + threadIdx.x;
  if (i >= n4) return;
  float4 f = ((const float4*)src)[i];
  bf16t* d = dst + i * 4;
  d[0] = (bf16t)f.x; d[1] = (bf16t)f.y; d[2] = (bf16t)f.z; d[3] = (bf16t)f.w;
}

// dst[(c)*dstLd + rowOff + r] = (bf16)src[r*2048 + c]
__global__ __launch_bounds__(256) void cvt_transpose_kernel(
    const float* __restrict__ src, bf16t* __restrict__ dst,
    long dstLd, int rowOff)
{
  __shared__ bf16t tile[32][33];
  const int bx = blockIdx.x * 32;
  const int by = blockIdx.y * 32;
  const int tx = threadIdx.x & 31, ty = threadIdx.x >> 5;
  #pragma unroll
  for (int j = 0; j < 32; j += 8)
    tile[ty + j][tx] = (bf16t)src[(size_t)(by + ty + j) * Dn + bx + tx];
  __syncthreads();
  #pragma unroll
  for (int j = 0; j < 32; j += 8)
    dst[(size_t)(bx + ty + j) * dstLd + rowOff + by + tx] = tile[tx][ty + j];
}

// ---------------------------------------------------------------------------
// GEMM: C = (A @ Bt^T + bias) * alpha.  A bf16 (M x K) row-major,
// Bt bf16 (N x K) row-major.  128x128 C tile, 8 waves (4x2), 32x64 per wave.
// K staged 64 deep, DOUBLE-BUFFERED async LDS staging (8 chunks/thread/tile).
// ---------------------------------------------------------------------------
template <bool OUT_F32, bool OUT_TRANS>
__global__ __launch_bounds__(256) void gemm_bf16_wmma_kernel(
    const bf16t* __restrict__ A, const bf16t* __restrict__ Bt,
    const float* __restrict__ bias, void* __restrict__ Cout,
    long outLd, int rowOff, float alpha)
{
  __shared__ __align__(16) bf16t As[2][128 * 72];   // 128 x 64 (+8 pad)
  __shared__ __align__(16) bf16t Bs[2][128 * 72];
  const int t = threadIdx.x, lane = t & 31, wave = t >> 5;
  const int wm = wave >> 1, wn = wave & 1;
  const int m0 = blockIdx.y * 128, n0 = blockIdx.x * 128;

  auto stage = [&](int k0, int buf) {
    #pragma unroll
    for (int it = 0; it < 4; ++it) {
      int idx = t + it * 256;                 // 1024 chunks per tile
      int r = idx >> 3, c8 = (idx & 7) * 8;
      copy16_g2l(A  + (size_t)(m0 + r) * Dn + k0 + c8, &As[buf][r * 72 + c8]);
      copy16_g2l(Bt + (size_t)(n0 + r) * Dn + k0 + c8, &Bs[buf][r * 72 + c8]);
    }
  };

  v8f acc[2][4] = {};
  const int NK = Dn / 64;                     // 32 K-tiles
  stage(0, 0);

  for (int i = 0; i < NK; ++i) {
    if (i + 1 < NK) { stage((i + 1) * 64, (i + 1) & 1); async_wait_prev8(); }
    else            { async_join(); }
    __syncthreads();

    const bf16t* Ab = As[i & 1];
    const bf16t* Bb = Bs[i & 1];
    #pragma unroll
    for (int ks = 0; ks < 2; ++ks) {
      v16bf af[2], bfr[4];
      #pragma unroll
      for (int mt = 0; mt < 2; ++mt)
        af[mt] = load_a_frag(&Ab[(wm * 32 + mt * 16) * 72 + ks * 32], 72, lane);
      #pragma unroll
      for (int nt = 0; nt < 4; ++nt)
        bfr[nt] = load_b_frag(&Bb[(wn * 64 + nt * 16) * 72 + ks * 32], 72, lane);
      #pragma unroll
      for (int mt = 0; mt < 2; ++mt)
        #pragma unroll
        for (int nt = 0; nt < 4; ++nt)
          acc[mt][nt] = wmma_bf16(af[mt], bfr[nt], acc[mt][nt]);
    }
    __syncthreads();
  }

  const int lg = lane >> 4, ln = lane & 15;
  #pragma unroll
  for (int mt = 0; mt < 2; ++mt)
    #pragma unroll
    for (int nt = 0; nt < 4; ++nt) {
      int col = n0 + wn * 64 + nt * 16 + ln;
      float bb = bias[col];
      #pragma unroll
      for (int i = 0; i < 8; ++i) {
        int row = m0 + wm * 32 + mt * 16 + lg * 8 + i;
        float v = (acc[mt][nt][i] + bb) * alpha;
        size_t off = OUT_TRANS ? ((size_t)col * outLd + rowOff + row)
                               : ((size_t)row * outLd + col);
        if (OUT_F32) ((float*)Cout)[off] = v;
        else         ((bf16t*)Cout)[off] = (bf16t)v;
      }
    }
}

// ---------------------------------------------------------------------------
// Flash attention, split prefix/suffix softmax, double-buffered async K/V
// staging. Gated prefix result is flushed to `ob` at the prefix/suffix
// boundary (kb==7) and combined via global RMW at the end (saves 64 VGPRs).
// ---------------------------------------------------------------------------
__global__ __launch_bounds__(256) void attn_flash_wmma_kernel(
    const bf16t* __restrict__ qb, const bf16t* __restrict__ kf,
    const bf16t* __restrict__ vT, const float* __restrict__ gate,
    bf16t* __restrict__ ob)
{
  __shared__ __align__(16) bf16t Qs[128 * 136];
  __shared__ __align__(16) bf16t Ks[2][64 * 136];
  __shared__ __align__(16) bf16t Vt[2][128 * 72];
  __shared__ __align__(16) bf16t Ps[8][16 * 72];

  const int h = blockIdx.y, qt = blockIdx.x;
  const int qbase = qt * 128;
  const int t = threadIdx.x, lane = t & 31, wave = t >> 5;
  const int lg = lane >> 4, ln = lane & 15;

  // stage Q tile (async, 8 chunks/thread)
  #pragma unroll
  for (int it = 0; it < 8; ++it) {
    int idx = t + it * 256;
    int r = idx >> 4, c8 = (idx & 15) * 8;
    copy16_g2l(qb + (size_t)(qbase + r) * Dn + h * HDn + c8, &Qs[r * 136 + c8]);
  }

  auto stageKV = [&](int kb, int buf) {
    #pragma unroll
    for (int it = 0; it < 4; ++it) {
      int idx = t + it * 256;
      int r = idx >> 4, c8 = (idx & 15) * 8;
      copy16_g2l(kf + (size_t)(kb * 64 + r) * Dn + h * HDn + c8,
                 &Ks[buf][r * 136 + c8]);
      int hd = idx >> 3, c82 = (idx & 7) * 8;
      copy16_g2l(vT + (size_t)(h * HDn + hd) * KVn + kb * 64 + c82,
                 &Vt[buf][hd * 72 + c82]);
    }
  };

  float mrun[8], lrun[8];
  v8f oacc[8] = {};
  #pragma unroll
  for (int i = 0; i < 8; ++i) { mrun[i] = -3.0e38f; lrun[i] = 0.0f; }

  const int nb = 8 + 2 * qt + 2;
  stageKV(0, 0);

  for (int kb = 0; kb < nb; ++kb) {
    if (kb + 1 < nb) { stageKV(kb + 1, (kb + 1) & 1); async_wait_prev8(); }
    else             { async_join(); }
    __syncthreads();

    const bf16t* Kb = Ks[kb & 1];
    const bf16t* Vb = Vt[kb & 1];

    // S(16x64) = Q_wave @ K^T
    v8f sacc[4] = {};
    #pragma unroll
    for (int ks = 0; ks < 4; ++ks) {
      v16bf a = load_a_frag(&Qs[(wave * 16) * 136 + ks * 32], 136, lane);
      #pragma unroll
      for (int nt = 0; nt < 4; ++nt) {
        v16bf b = load_b_frag(&Kb[(nt * 16) * 136 + ks * 32], 136, lane);
        sacc[nt] = wmma_bf16(a, b, sacc[nt]);
      }
    }

    // causal mask (suffix only) + per-row max
    const bool suf = (kb >= 8);
    float rmax[8];
    #pragma unroll
    for (int i = 0; i < 8; ++i) rmax[i] = -3.0e38f;
    #pragma unroll
    for (int nt = 0; nt < 4; ++nt) {
      int j = kb * 64 + nt * 16 + ln;
      #pragma unroll
      for (int i = 0; i < 8; ++i) {
        float s = sacc[nt][i];
        if (suf) {
          int qi = qbase + wave * 16 + lg * 8 + i;
          if (j - Pn > qi) { s = -1.0e30f; sacc[nt][i] = s; }
        }
        rmax[i] = fmaxf(rmax[i], s);
      }
    }
    #pragma unroll
    for (int i = 0; i < 8; ++i)
      #pragma unroll
      for (int off = 1; off < 16; off <<= 1)
        rmax[i] = fmaxf(rmax[i], __shfl_xor(rmax[i], off, 32));

    // online softmax
    float rsum[8];
    #pragma unroll
    for (int i = 0; i < 8; ++i) {
      float nm = fmaxf(mrun[i], rmax[i]);
      float sc = __expf(mrun[i] - nm);
      mrun[i] = nm;
      lrun[i] *= sc;
      #pragma unroll
      for (int ht = 0; ht < 8; ++ht) oacc[ht][i] *= sc;
      float rs = 0.0f;
      #pragma unroll
      for (int nt = 0; nt < 4; ++nt) {
        float p = __expf(sacc[nt][i] - nm);
        sacc[nt][i] = p;
        rs += p;
      }
      rsum[i] = rs;
    }
    #pragma unroll
    for (int i = 0; i < 8; ++i) {
      #pragma unroll
      for (int off = 1; off < 16; off <<= 1)
        rsum[i] += __shfl_xor(rsum[i], off, 32);
      lrun[i] += rsum[i];
    }

    // P (C-layout) -> per-wave LDS (A-layout source), then O += P @ V
    bf16t* pw = Ps[wave];
    #pragma unroll
    for (int nt = 0; nt < 4; ++nt)
      #pragma unroll
      for (int i = 0; i < 8; ++i)
        pw[(lg * 8 + i) * 72 + nt * 16 + ln] = (bf16t)sacc[nt][i];

    #pragma unroll
    for (int ks = 0; ks < 2; ++ks) {
      v16bf a = load_a_frag(pw + ks * 32, 72, lane);
      #pragma unroll
      for (int ht = 0; ht < 8; ++ht) {
        v16bf b = load_b_frag(&Vb[(ht * 16) * 72 + ks * 32], 72, lane);
        oacc[ht] = wmma_bf16(a, b, oacc[ht]);
      }
    }

    if (kb == 7) {   // flush gated prefix result to global, restart stats
      float g = gate[h];
      v8f zf = {};
      #pragma unroll
      for (int ht = 0; ht < 8; ++ht) {
        int col = h * HDn + ht * 16 + ln;
        #pragma unroll
        for (int i = 0; i < 8; ++i) {
          int row = qbase + wave * 16 + lg * 8 + i;
          ob[(size_t)row * Dn + col] = (bf16t)(oacc[ht][i] * (g / lrun[i]));
        }
        oacc[ht] = zf;
      }
      #pragma unroll
      for (int i = 0; i < 8; ++i) { mrun[i] = -3.0e38f; lrun[i] = 0.0f; }
    }
    __syncthreads();
  }

  // combine: o = prefix_part (in ob) + suffix_part / l
  #pragma unroll
  for (int ht = 0; ht < 8; ++ht) {
    int col = h * HDn + ht * 16 + ln;
    #pragma unroll
    for (int i = 0; i < 8; ++i) {
      int row = qbase + wave * 16 + lg * 8 + i;
      size_t off = (size_t)row * Dn + col;
      float v = (float)ob[off] + oacc[ht][i] * (1.0f / lrun[i]);
      ob[off] = (bf16t)v;
    }
  }
}

// ---------------------------------------------------------------------------
extern "C" void kernel_launch(void* const* d_in, const int* in_sizes, int n_in,
                              void* d_out, int out_size, void* d_ws, size_t ws_size,
                              hipStream_t stream) {
  (void)in_sizes; (void)n_in; (void)out_size; (void)ws_size;
  const float* x    = (const float*)d_in[0];
  const float* pk   = (const float*)d_in[1];
  const float* pv   = (const float*)d_in[2];
  const float* Wq   = (const float*)d_in[3];
  const float* bq   = (const float*)d_in[4];
  const float* Wk   = (const float*)d_in[5];
  const float* bk   = (const float*)d_in[6];
  const float* Wv   = (const float*)d_in[7];
  const float* bv   = (const float*)d_in[8];
  const float* gate = (const float*)d_in[9];
  const float* Wout = (const float*)d_in[10];
  const float* bout = (const float*)d_in[11];
  float* out = (float*)d_out;

  // workspace (bf16): x | WqT | WkT | WvT | WoutT | q | kfull | vT | o
  const size_t SQ = (size_t)Qn * Dn;
  const size_t SK = (size_t)KVn * Dn;
  bf16t* xb    = (bf16t*)d_ws;
  bf16t* WqT   = xb    + SQ;
  bf16t* WkT   = WqT   + SQ;
  bf16t* WvT   = WkT   + SQ;
  bf16t* WoT   = WvT   + SQ;
  bf16t* qb    = WoT   + SQ;
  bf16t* kfull = qb    + SQ;
  bf16t* vTb   = kfull + SK;             // [2048, 2560] hd-major
  bf16t* obuf  = vTb   + SK;

  dim3 blk(256);
  cvt_flat_kernel<<<dim3((SQ / 4 + 255) / 256), blk, 0, stream>>>(x, xb, SQ / 4);
  cvt_flat_kernel<<<dim3(((size_t)Pn * Dn / 4 + 255) / 256), blk, 0, stream>>>(
      pk, kfull, (size_t)Pn * Dn / 4);
  dim3 gt(Dn / 32, Dn / 32);
  cvt_transpose_kernel<<<gt, blk, 0, stream>>>(Wq,   WqT, Dn, 0);
  cvt_transpose_kernel<<<gt, blk, 0, stream>>>(Wk,   WkT, Dn, 0);
  cvt_transpose_kernel<<<gt, blk, 0, stream>>>(Wv,   WvT, Dn, 0);
  cvt_transpose_kernel<<<gt, blk, 0, stream>>>(Wout, WoT, Dn, 0);
  cvt_transpose_kernel<<<dim3(Dn / 32, Pn / 32), blk, 0, stream>>>(pv, vTb, KVn, 0);

  dim3 gg(Dn / 128, Qn / 128);
  const float qscale = 0.088388347648318447f;   // 1/sqrt(HD)
  gemm_bf16_wmma_kernel<false, false><<<gg, blk, 0, stream>>>(
      xb, WqT, bq, qb, (long)Dn, 0, qscale);
  gemm_bf16_wmma_kernel<false, false><<<gg, blk, 0, stream>>>(
      xb, WkT, bk, kfull + (size_t)Pn * Dn, (long)Dn, 0, 1.0f);
  gemm_bf16_wmma_kernel<false, true><<<gg, blk, 0, stream>>>(
      xb, WvT, bv, vTb, (long)KVn, Pn, 1.0f);

  attn_flash_wmma_kernel<<<dim3(Qn / 128, Hn), blk, 0, stream>>>(
      qb, kfull, vTb, gate, obuf);

  gemm_bf16_wmma_kernel<true, false><<<gg, blk, 0, stream>>>(
      obuf, WoT, bout, out, (long)Dn, 0, 1.0f);
}
